// GCN_7885559956092
// MI455X (gfx1250) — matrix operand order repaired and checked by reading, not compiled, not run on hardware
//
#include <hip/hip_runtime.h>

typedef __attribute__((ext_vector_type(16))) _Float16 v16h;
typedef __attribute__((ext_vector_type(8)))  float    v8f;

#define F_IN 256
#define HID  64
#define CLS  40
#define CLS_PAD 48

// ---------------------------------------------------------------------------
// Degree / normalization: deg = 1 (self loop) + #incoming edges; dinv = rsqrt
// ---------------------------------------------------------------------------
__global__ void k_init_deg(float* __restrict__ deg, int N) {
    int i = blockIdx.x * blockDim.x + threadIdx.x;
    if (i < N) deg[i] = 1.0f;
}

__global__ void k_count_deg(const int* __restrict__ dst, float* __restrict__ deg, int E) {
    int e = blockIdx.x * blockDim.x + threadIdx.x;
    if (e < E) atomicAdd(&deg[dst[e]], 1.0f);
}

__global__ void k_finalize_dinv(float* __restrict__ deg, int N) {
    int i = blockIdx.x * blockDim.x + threadIdx.x;
    if (i < N) deg[i] = rsqrtf(deg[i]);   // deg >= 1 always
}

// Per-edge symmetric norm, computed once: w[e] = dinv[src]*dinv[dst]
__global__ void k_edge_norm(const int* __restrict__ src, const int* __restrict__ dst,
                            const float* __restrict__ dinv, float* __restrict__ w, int E) {
    int e = blockIdx.x * blockDim.x + threadIdx.x;
    if (e < E) w[e] = dinv[src[e]] * dinv[dst[e]];
}

// ---------------------------------------------------------------------------
// GEMM1: h1[N,64] = x[N,256] @ W1[256,64], f16 WMMA / f32 accumulate.
// One wave owns a 16-row stripe of x (read once) and computes all 4 column
// tiles (4 accumulators, A-fragment reused across 4 back-to-back WMMAs).
// W1 is staged in LDS pre-swizzled into B-fragment order so each lane's
// 16 halves are contiguous -> ds_load_b128 instead of 16x ds_load_u16.
//   Bsw[((kg*4 + n0)*32 + lane)*16 + h] = W1_f16[k, n]
//     k = kg*32 + (lane>>4)*16 + h,  n = n0*16 + (lane&15)
// (contiguous k order, same as the A fragment below; wave32)
// ---------------------------------------------------------------------------
__global__ __launch_bounds__(128)
void k_gemm1_wmma(const float* __restrict__ x, const float* __restrict__ W1,
                  float* __restrict__ h1, int N) {
    __shared__ __align__(32) _Float16 Bsw[F_IN * HID];   // 32 KB of 320 KB/WGP
    for (int t = threadIdx.x; t < F_IN * HID; t += 128) {
        const int h    = t & 15;
        const int ln   = (t >> 4) & 31;
        const int n0   = (t >> 9) & 3;
        const int kg   = t >> 11;                        // 0..7
        const int k    = (kg << 5) + ((ln >> 4) << 4) + h;
        const int n    = (n0 << 4) + (ln & 15);
        Bsw[t] = (_Float16)W1[(size_t)k * HID + n];
    }
    __syncthreads();

    const int lane    = threadIdx.x & 31;
    const int wave    = threadIdx.x >> 5;
    const int halfSel = lane >> 4;                // 0: lanes 0-15, 1: lanes 16-31
    const int mr      = lane & 15;
    const int mbase   = blockIdx.x * 64 + wave * 16;

    int arow = mbase + mr;
    if (arow >= N) arow = N - 1;                  // clamp: tail rows give garbage
    const float* xrow = x + (size_t)arow * F_IN;  // in rows we never store

    v8f c[4] = {};
    for (int kk = 0; kk < F_IN; kk += 32) {
        // A fragment (16x32 f16): lanes 0-15 hold K=kk..+15;
        // lanes 16-31 hold K=kk+16..+31 for the same row.
        v16h a;
        const int ka = kk + halfSel * 16;
        #pragma unroll
        for (int h = 0; h < 8; ++h) {
            a[h]     = (_Float16)xrow[ka + h];
            a[h + 8] = (_Float16)xrow[ka + 8 + h];
        }
        const int kg = kk >> 5;
        #pragma unroll
        for (int n0 = 0; n0 < 4; ++n0) {
            const v16h b = *(const v16h*)&Bsw[(((kg << 2) + n0) * 32 + lane) << 4];
            c[n0] = __builtin_amdgcn_wmma_f32_16x16x32_f16(false, a, false, b,
                                                           (short)0, c[n0], false, false);
        }
    }
    // D layout: VGPR r -> row mbase + r + 8*halfSel, col n0*16 + mr
    #pragma unroll
    for (int r = 0; r < 8; ++r) {
        const int row = mbase + r + halfSel * 8;
        if (row < N) {
            #pragma unroll
            for (int n0 = 0; n0 < 4; ++n0)
                h1[(size_t)row * HID + n0 * 16 + mr] = c[n0][r];
        }
    }
}

// ---------------------------------------------------------------------------
// GEMM2: h2[N,48pad] = hrelu[N,64] @ W2[64,40]. One wave: 16 rows x 48 cols
// (3 accumulators). W2 staged in LDS pre-swizzled + zero-padded 40 -> 48.
//   Bsw[((n0*2 + kg)*32 + lane)*16 + h],  n0 in 0..2, kg in 0..1
// ---------------------------------------------------------------------------
__global__ __launch_bounds__(128)
void k_gemm2_wmma(const float* __restrict__ h, const float* __restrict__ W2,
                  float* __restrict__ h2, int N) {
    __shared__ __align__(32) _Float16 Bsw[HID * CLS_PAD];   // 6 KB
    for (int t = threadIdx.x; t < HID * CLS_PAD; t += 128) {
        const int hh = t & 15;
        const int ln = (t >> 4) & 31;
        const int kg = (t >> 9) & 1;
        const int n0 = t >> 10;                              // 0..2
        const int k  = (kg << 5) + ((ln >> 4) << 4) + hh;
        const int n  = (n0 << 4) + (ln & 15);
        Bsw[t] = (n < CLS) ? (_Float16)W2[(size_t)k * CLS + n] : (_Float16)0.0f;
    }
    __syncthreads();

    const int lane    = threadIdx.x & 31;
    const int wave    = threadIdx.x >> 5;
    const int halfSel = lane >> 4;
    const int mr      = lane & 15;
    const int mbase   = blockIdx.x * 64 + wave * 16;

    int arow = mbase + mr;
    if (arow >= N) arow = N - 1;
    const float* hrow = h + (size_t)arow * HID;

    v8f c[3] = {};
    for (int kk = 0; kk < HID; kk += 32) {
        v16h a;
        const int ka = kk + halfSel * 16;
        #pragma unroll
        for (int hh = 0; hh < 8; ++hh) {
            a[hh]     = (_Float16)hrow[ka + hh];
            a[hh + 8] = (_Float16)hrow[ka + 8 + hh];
        }
        const int kg = kk >> 5;
        #pragma unroll
        for (int n0 = 0; n0 < 3; ++n0) {
            const v16h b = *(const v16h*)&Bsw[((n0 * 2 + kg) * 32 + lane) << 4];
            c[n0] = __builtin_amdgcn_wmma_f32_16x16x32_f16(false, a, false, b,
                                                           (short)0, c[n0], false, false);
        }
    }
    #pragma unroll
    for (int r = 0; r < 8; ++r) {
        const int row = mbase + r + halfSel * 8;
        if (row < N) {
            #pragma unroll
            for (int n0 = 0; n0 < 3; ++n0)
                h2[(size_t)row * CLS_PAD + n0 * 16 + mr] = c[n0][r];
        }
    }
}

// ---------------------------------------------------------------------------
// Aggregation layer 1 (feature dim 64): self-loop init, edge scatter, bias+ReLU
// ---------------------------------------------------------------------------
__global__ void k_self_init64(const float* __restrict__ h, const float* __restrict__ dinv,
                              float* __restrict__ agg, int N) {
    int idx = blockIdx.x * blockDim.x + threadIdx.x;   // i*64 + j
    int i = idx >> 6;
    if (i < N) {
        float di = dinv[i];
        agg[idx] = h[idx] * di * di;
    }
}

__global__ void k_scatter64(const int* __restrict__ src, const int* __restrict__ dst,
                            const float* __restrict__ w, const float* __restrict__ h,
                            float* __restrict__ agg, int E) {
    int t = blockIdx.x * blockDim.x + threadIdx.x;     // e*64 + j (E*64 < 2^31)
    int e = t >> 6;
    int j = t & 63;
    if (e < E) {
        int s = src[e], d = dst[e];                    // wave-uniform broadcasts
        atomicAdd(&agg[(size_t)d * HID + j], h[(size_t)s * HID + j] * w[e]);
    }
}

__global__ void k_bias_relu64(float* __restrict__ agg, const float* __restrict__ b1, int N) {
    int idx = blockIdx.x * blockDim.x + threadIdx.x;
    int i = idx >> 6;
    int j = idx & 63;
    if (i < N) {
        float v = agg[idx] + b1[j];
        agg[idx] = v > 0.0f ? v : 0.0f;
    }
}

// ---------------------------------------------------------------------------
// Aggregation layer 2 (feature dim 40, h2 padded to 48): writes into d_out
// ---------------------------------------------------------------------------
__global__ void k_self_init40(const float* __restrict__ h2, const float* __restrict__ dinv,
                              float* __restrict__ out, int N) {
    int idx = blockIdx.x * blockDim.x + threadIdx.x;   // i*64 + j groups
    int i = idx >> 6;
    int j = idx & 63;
    if (i < N && j < CLS) {
        float di = dinv[i];
        out[(size_t)i * CLS + j] = h2[(size_t)i * CLS_PAD + j] * di * di;
    }
}

__global__ void k_scatter40(const int* __restrict__ src, const int* __restrict__ dst,
                            const float* __restrict__ w, const float* __restrict__ h2,
                            float* __restrict__ out, int E) {
    int t = blockIdx.x * blockDim.x + threadIdx.x;
    int e = t >> 6;
    int j = t & 63;
    if (e < E && j < CLS) {
        int s = src[e], d = dst[e];
        atomicAdd(&out[(size_t)d * CLS + j], h2[(size_t)s * CLS_PAD + j] * w[e]);
    }
}

__global__ void k_bias40(float* __restrict__ out, const float* __restrict__ b2, int N) {
    int idx = blockIdx.x * blockDim.x + threadIdx.x;
    int i = idx >> 6;
    int j = idx & 63;
    if (i < N && j < CLS) out[(size_t)i * CLS + j] += b2[j];
}

// ---------------------------------------------------------------------------
// Host-side launch
// ---------------------------------------------------------------------------
extern "C" void kernel_launch(void* const* d_in, const int* in_sizes, int n_in,
                              void* d_out, int out_size, void* d_ws, size_t ws_size,
                              hipStream_t stream) {
    const float* x     = (const float*)d_in[0];
    const float* W1    = (const float*)d_in[1];
    const float* b1    = (const float*)d_in[2];
    const float* W2    = (const float*)d_in[3];
    const float* b2    = (const float*)d_in[4];
    const int*   edges = (const int*)d_in[5];

    const int N = in_sizes[0] / F_IN;       // 100000
    const int E = in_sizes[5] / 2;          // 1600000
    const int* src = edges;
    const int* dst = edges + E;

    float* out = (float*)d_out;             // [N, 40]

    // workspace (floats): dinv[N] | h1[N*64] | agg1[N*64] | h2[N*48] | w[E]
    float* dinv = (float*)d_ws;
    float* h1   = dinv + N;
    float* agg1 = h1 + (size_t)N * HID;
    float* h2   = agg1 + (size_t)N * HID;
    float* wE   = h2 + (size_t)N * CLS_PAD;

    const int B = 256;
    const int gN    = (N + B - 1) / B;
    const int gE    = (E + B - 1) / B;
    const int gN64  = (int)(((size_t)N * 64 + B - 1) / B);
    const int gE64  = (int)(((size_t)E * 64 + B - 1) / B);
    const int gTile = (N + 63) / 64;        // 1563, tail-guarded

    // normalization (computed once, shared by both layers)
    k_init_deg<<<gN, B, 0, stream>>>(dinv, N);
    k_count_deg<<<gE, B, 0, stream>>>(dst, dinv, E);
    k_finalize_dinv<<<gN, B, 0, stream>>>(dinv, N);
    k_edge_norm<<<gE, B, 0, stream>>>(src, dst, dinv, wE, E);

    // layer 1
    k_gemm1_wmma<<<gTile, 128, 0, stream>>>(x, W1, h1, N);
    k_self_init64<<<gN64, B, 0, stream>>>(h1, dinv, agg1, N);
    k_scatter64<<<gE64, B, 0, stream>>>(src, dst, wE, h1, agg1, E);
    k_bias_relu64<<<gN64, B, 0, stream>>>(agg1, b1, N);

    // layer 2
    k_gemm2_wmma<<<gTile, 128, 0, stream>>>(agg1, W2, h2, N);
    k_self_init40<<<gN64, B, 0, stream>>>(h2, dinv, out, N);
    k_scatter40<<<gE64, B, 0, stream>>>(src, dst, wE, h2, out, E);
    k_bias40<<<gN64, B, 0, stream>>>(out, b2, N);
}
